// ModeloLinguagemTransformer_28630251995760
// MI455X (gfx1250) — compile-verified, matrix operand order checked by reading, not run on hardware
//
#include <hip/hip_runtime.h>
#include <math.h>

// ---------------- problem constants ----------------
constexpr int V   = 32000;
constexpr int D   = 1024;
constexpr int H   = 16;
constexpr int L   = 6;
constexpr int DFF = 4096;
constexpr int DH  = 64;
constexpr int B   = 2;
constexpr int S   = 1024;
constexpr int BS  = B * S;

// ---------------- types / helpers ----------------
typedef __bf16 bf16;
typedef bf16  v16bf __attribute__((ext_vector_type(16)));
typedef bf16  v8bf  __attribute__((ext_vector_type(8)));
typedef float v8f   __attribute__((ext_vector_type(8)));
typedef unsigned int u32x4 __attribute__((ext_vector_type(4)));
typedef int  i32x4 __attribute__((ext_vector_type(4)));
typedef int  i32x8 __attribute__((ext_vector_type(8)));

union frag16 { v16bf v; v8bf h[2]; };

__device__ __forceinline__ bf16 f2bf(float f) {
    unsigned u = __builtin_bit_cast(unsigned, f);
    u += 0x7fffu + ((u >> 16) & 1u);
    unsigned short h = (unsigned short)(u >> 16);
    return __builtin_bit_cast(bf16, h);
}
__device__ __forceinline__ float bf2f(bf16 b) {
    unsigned u = ((unsigned)__builtin_bit_cast(unsigned short, b)) << 16;
    return __builtin_bit_cast(float, u);
}
__device__ __forceinline__ v8f wmma_bf16(v16bf a, v16bf b, v8f c) {
    return __builtin_amdgcn_wmma_f32_16x16x32_bf16(
        false, a, false, b, (short)0, c, false, false);
}
// wave-local LDS ordering: LDS ops of one wave execute in order at the LDS unit;
// this only needs to stop compile-time reordering and drain DScnt.
__device__ __forceinline__ void wave_lds_fence() {
    asm volatile("s_wait_dscnt 0x0" ::: "memory");
}

// ---------------- TDM: 2D tile (32 elems x 128 rows, bf16) global -> LDS ----------------
// LDS layout: 128 rows, 64B data + 16B pad -> 80B row stride (16B aligned, conflict-free).
// D# per ISA 8.3/8.4: count=1, type=2, data_size=1(2B), pad_enable, pad_interval=16dw(code3),
// pad_amount=4dw(code3), tile_dim0=32, tile_dim1=128.
__device__ __forceinline__ void tdm_load_tile(unsigned ldsOff, const bf16* gptr,
                                              int rowLen /*elements*/, int nRows)
{
    unsigned long long ga = (unsigned long long)(uintptr_t)gptr;
    u32x4 g0;
    g0[0] = 1u;                                   // count=1, user mode
    g0[1] = ldsOff;                               // lds_addr
    g0[2] = (unsigned)(ga & 0xffffffffu);         // global_addr[31:0]
    g0[3] = ((unsigned)(ga >> 32) & 0x01ffffffu) | 0x80000000u; // addr[56:32] | type=2
    i32x8 g1;
    g1[0] = (int)0x06D10000u;                     // data_size=1, pad_en, interval=3, amount=3
    g1[1] = (int)(((unsigned)rowLen & 0xffffu) << 16);                // tensor_dim0 lo
    g1[2] = (int)((((unsigned)rowLen >> 16) & 0xffffu) |
                  (((unsigned)nRows & 0xffffu) << 16));               // dim0 hi | dim1 lo
    g1[3] = (int)((((unsigned)nRows >> 16) & 0xffffu) | (32u << 16)); // dim1 hi | tile_dim0=32
    g1[4] = (int)128u;                            // tile_dim1=128
    g1[5] = rowLen;                               // tensor_dim0_stride[31:0]
    g1[6] = 0;
    g1[7] = 0;
    i32x4 z4 = {0, 0, 0, 0};
#if __clang_major__ >= 23
    i32x8 z8 = {0, 0, 0, 0, 0, 0, 0, 0};
    __builtin_amdgcn_tensor_load_to_lds(g0, g1, z4, z4, z8, 0);
#else
    __builtin_amdgcn_tensor_load_to_lds(g0, g1, z4, z4, 0);
#endif
}

// ---------------- embedding ----------------
__global__ __launch_bounds__(256) void embed_kernel(
    const int* __restrict__ tokens, const float* __restrict__ emb_tok,
    const float* __restrict__ emb_pos, float* __restrict__ x)
{
    int idx = blockIdx.x * 256 + threadIdx.x;
    int d  = idx & (D - 1);
    int bs = idx >> 10;
    int s  = bs & (S - 1);
    int tok = tokens[bs];
    x[idx] = emb_tok[(size_t)tok * D + d] + emb_pos[(size_t)s * D + d];
}

// ---------------- layer norm: fp32 in -> bf16 out ----------------
__global__ __launch_bounds__(256) void layernorm_kernel(
    const float* __restrict__ x, const float* __restrict__ w,
    const float* __restrict__ b, bf16* __restrict__ out)
{
    __shared__ float red[16];
    const int row = blockIdx.x;
    const float* xr = x + (size_t)row * D;
    float s = 0.f, ss = 0.f;
    for (int i = threadIdx.x; i < D; i += 256) { float v = xr[i]; s += v; ss += v * v; }
    #pragma unroll
    for (int off = 16; off > 0; off >>= 1) {
        s  += __shfl_down(s,  off, 32);
        ss += __shfl_down(ss, off, 32);
    }
    int wave = threadIdx.x >> 5, lane = threadIdx.x & 31;
    if (lane == 0) { red[wave] = s; red[8 + wave] = ss; }
    __syncthreads();
    if (threadIdx.x == 0) {
        float ts = 0.f, tss = 0.f;
        for (int i = 0; i < 8; ++i) { ts += red[i]; tss += red[8 + i]; }
        red[0] = ts; red[8] = tss;
    }
    __syncthreads();
    float mu  = red[0] * (1.0f / D);
    float var = red[8] * (1.0f / D) - mu * mu;
    float inv = rsqrtf(var + 1e-5f);
    for (int i = threadIdx.x; i < D; i += 256)
        out[(size_t)row * D + i] = f2bf((xr[i] - mu) * inv * w[i] + b[i]);
}

// ---------------- weight transpose: fp32 RxC -> bf16 CxR ----------------
__global__ __launch_bounds__(256) void transpose_f32_bf16(
    const float* __restrict__ src, bf16* __restrict__ dst, int R, int C)
{
    __shared__ float tile[32][33];
    int tx = threadIdx.x & 31, ty = threadIdx.x >> 5;
    int rb = blockIdx.y * 32, cb = blockIdx.x * 32;
    #pragma unroll
    for (int p = 0; p < 4; ++p)
        tile[ty + p * 8][tx] = src[(size_t)(rb + ty + p * 8) * C + cb + tx];
    __syncthreads();
    #pragma unroll
    for (int p = 0; p < 4; ++p)
        dst[(size_t)(cb + ty + p * 8) * R + rb + tx] = f2bf(tile[tx][ty + p * 8]);
}

// ---------------- batched bf16 transpose (V -> V^T per batch) ----------------
__global__ __launch_bounds__(256) void transpose_bf16_batched(
    const bf16* __restrict__ src0, bf16* __restrict__ dst0, int R, int C)
{
    const bf16* src = src0 + (size_t)blockIdx.z * R * C;
    bf16*       dst = dst0 + (size_t)blockIdx.z * R * C;
    __shared__ bf16 tile[32][33];
    int tx = threadIdx.x & 31, ty = threadIdx.x >> 5;
    int rb = blockIdx.y * 32, cb = blockIdx.x * 32;
    #pragma unroll
    for (int p = 0; p < 4; ++p)
        tile[ty + p * 8][tx] = src[(size_t)(rb + ty + p * 8) * C + cb + tx];
    __syncthreads();
    #pragma unroll
    for (int p = 0; p < 4; ++p)
        dst[(size_t)(cb + ty + p * 8) * R + rb + tx] = tile[tx][ty + p * 8];
}

// ---------------- RoPE on bf16 q,k in place ----------------
__global__ __launch_bounds__(256) void rope_kernel(
    bf16* __restrict__ q, bf16* __restrict__ k)
{
    int p = blockIdx.x * 256 + threadIdx.x;
    int j   = p & (D / 2 - 1);
    int row = p >> 9;
    int s   = row & (S - 1);
    int head = j >> 5;
    int i    = j & 31;
    float theta = __powf(10000.0f, -((float)(2 * i)) * (1.0f / (float)DH));
    float ang = (float)s * theta;
    float c = cosf(ang), sn = sinf(ang);
    size_t o = (size_t)row * D + head * DH + 2 * i;
    float qr = bf2f(q[o]), qi = bf2f(q[o + 1]);
    q[o] = f2bf(qr * c - qi * sn);  q[o + 1] = f2bf(qr * sn + qi * c);
    float kr = bf2f(k[o]), ki = bf2f(k[o + 1]);
    k[o] = f2bf(kr * c - ki * sn);  k[o + 1] = f2bf(kr * sn + ki * c);
}

// ---------------- WMMA GEMM, double-buffered TDM staging ----------------
// C = act(A @ W + bias) [+resid];  A: MxK bf16 row-major, WT: NxK bf16.
// Block tile 128x128, K-step 32, 8 waves, 8 WMMAs / wave / K-step.
// Wave 0 issues TDM for step k+1 into the other buffer, then s_wait_tensorcnt(2)
// guarantees the current buffer (older in-order pair) is complete -> DMA overlaps WMMA.
constexpr int LROW = 40; // LDS row stride in bf16 elems (64B data + 16B TDM pad)

template<bool GELU, bool RESID, bool OUTBF>
__global__ __launch_bounds__(256) void gemm_kernel(
    const bf16* __restrict__ A, const bf16* __restrict__ WT,
    const float* __restrict__ bias, const float* __restrict__ resid,
    void* __restrict__ Cout, int M, int N, int K)
{
    __shared__ __align__(16) bf16 ldsA[2][128 * LROW];
    __shared__ __align__(16) bf16 ldsB[2][128 * LROW];

    const int t    = threadIdx.x;
    const int lane = t & 31;
    const int wave = t >> 5;
    const int mw   = wave & 3;
    const int nw   = wave >> 2;
    const int lg   = lane >> 4;
    const int cn   = lane & 15;
    const int rowBase = blockIdx.y * 128;
    const int colBase = blockIdx.x * 128;

    const unsigned aOff0 = (unsigned)(uintptr_t)&ldsA[0][0];
    const unsigned aOff1 = (unsigned)(uintptr_t)&ldsA[1][0];
    const unsigned bOff0 = (unsigned)(uintptr_t)&ldsB[0][0];
    const unsigned bOff1 = (unsigned)(uintptr_t)&ldsB[1][0];

    const bf16* Abase = A  + (size_t)rowBase * K;
    const bf16* Wbase = WT + (size_t)colBase * K;

    v8f acc[2][4] = {};

    if (t < 32) {                                  // prologue: stage k0=0 into buf 0
        tdm_load_tile(aOff0, Abase, K, M);
        tdm_load_tile(bOff0, Wbase, K, N);
    }

    int cur = 0;
    for (int k0 = 0; k0 < K; k0 += 32) {
        if (t < 32) {
            if (k0 + 32 < K) {                     // prefetch next step into other buffer
                tdm_load_tile(cur ? aOff0 : aOff1, Abase + k0 + 32, K, M);
                tdm_load_tile(cur ? bOff0 : bOff1, Wbase + k0 + 32, K, N);
                __builtin_amdgcn_s_wait_tensorcnt(2);   // oldest pair (current buf) done
            } else {
                __builtin_amdgcn_s_wait_tensorcnt(0);
            }
        }
        __syncthreads();                           // current tiles visible to all waves

        const bf16* lA = &ldsA[cur][0];
        const bf16* lB = &ldsB[cur][0];
        frag16 fa[2];
        #pragma unroll
        for (int ms = 0; ms < 2; ++ms) {
            const bf16* ar = lA + (size_t)((mw * 2 + ms) * 16 + cn) * LROW;
            fa[ms].h[0] = *(const v8bf*)(ar + lg * 8);
            fa[ms].h[1] = *(const v8bf*)(ar + 16 + lg * 8);
        }
        #pragma unroll
        for (int j = 0; j < 4; ++j) {
            const bf16* br = lB + (size_t)((nw * 4 + j) * 16 + cn) * LROW;
            frag16 fb;
            fb.h[0] = *(const v8bf*)(br + lg * 8);
            fb.h[1] = *(const v8bf*)(br + 16 + lg * 8);
            acc[0][j] = wmma_bf16(fa[0].v, fb.v, acc[0][j]);
            acc[1][j] = wmma_bf16(fa[1].v, fb.v, acc[1][j]);
        }
        __syncthreads();                           // reads done before buf is re-filled
        cur ^= 1;
    }

    // epilogue: C layout -> row = base + r (lanes 0-15) / base + 8 + r (lanes 16-31)
    #pragma unroll
    for (int ms = 0; ms < 2; ++ms) {
        #pragma unroll
        for (int j = 0; j < 4; ++j) {
            int col = colBase + (nw * 4 + j) * 16 + cn;
            float bb = bias[col];
            #pragma unroll
            for (int r = 0; r < 8; ++r) {
                int row = rowBase + (mw * 2 + ms) * 16 + lg * 8 + r;
                float o = acc[ms][j][r] + bb;
                if (GELU)  o = 0.5f * o * (1.0f + erff(o * 0.7071067811865475f));
                if (RESID) o += resid[(size_t)row * N + col];
                if (OUTBF) ((bf16*)Cout)[(size_t)row * N + col] = f2bf(o);
                else       ((float*)Cout)[(size_t)row * N + col] = o;
            }
        }
    }
}

// ---------------- flash attention: 128 threads = 4 independent q-tiles ----------------
// No block barriers (per-wave trip counts differ); P restaging is intra-wave only,
// ordered by wave_lds_fence().
__global__ __launch_bounds__(128) void attn_kernel(
    const bf16* __restrict__ q, const bf16* __restrict__ k,
    const bf16* __restrict__ vt, bf16* __restrict__ o)
{
    __shared__ __align__(32) bf16 plds[4][32][16];

    const int lane = threadIdx.x & 31;
    const int w    = threadIdx.x >> 5;
    const int lg = lane >> 4, cn = lane & 15;
    const int qt   = blockIdx.x * 4 + w;
    const int head = blockIdx.y;
    const int b    = blockIdx.z;
    const int qBase = qt * 16;
    bf16 (* const pl)[16] = plds[w];

    // Q fragments: direct global b128 loads (row-major bf16 is fragment-native)
    const bf16* qrow = q + ((size_t)(b * S + qBase + cn)) * D + head * DH;
    frag16 aQ[2];
    #pragma unroll
    for (int f = 0; f < 2; ++f) {
        aQ[f].h[0] = *(const v8bf*)(qrow + f * 32 + lg * 8);
        aQ[f].h[1] = *(const v8bf*)(qrow + f * 32 + 16 + lg * 8);
    }

    float mrow[8], lrow[8];
    #pragma unroll
    for (int j = 0; j < 8; ++j) { mrow[j] = -3.0e38f; lrow[j] = 0.f; }
    v8f oacc[4] = {};

    const int ktEnd = (qBase + 15) >> 5;
    for (int kt = 0; kt <= ktEnd; ++kt) {
        const int kBase = kt * 32;

        // scores: QK^T, B-fragments straight from K rows (k-dim = dh contiguous)
        v8f s0 = {}, s1 = {};
        #pragma unroll
        for (int ktt = 0; ktt < 2; ++ktt) {
            const bf16* krow = k + ((size_t)(b * S + kBase + ktt * 16 + cn)) * D + head * DH;
            #pragma unroll
            for (int f = 0; f < 2; ++f) {
                frag16 fb;
                fb.h[0] = *(const v8bf*)(krow + f * 32 + lg * 8);
                fb.h[1] = *(const v8bf*)(krow + f * 32 + 16 + lg * 8);
                if (ktt == 0) s0 = wmma_bf16(aQ[f].v, fb.v, s0);
                else          s1 = wmma_bf16(aQ[f].v, fb.v, s1);
            }
        }

        wave_lds_fence();   // previous iteration's aP loads complete before restore
        // causal mask + 1/sqrt(DH) scale + online softmax
        #pragma unroll
        for (int j = 0; j < 8; ++j) {
            int row = qBase + j + lg * 8;
            float x0 = (kBase + cn      <= row) ? s0[j] * 0.125f : -1.0e9f;
            float x1 = (kBase + 16 + cn <= row) ? s1[j] * 0.125f : -1.0e9f;
            float mx = fmaxf(x0, x1);
            #pragma unroll
            for (int off = 8; off > 0; off >>= 1)
                mx = fmaxf(mx, __shfl_xor(mx, off, 16));
            float mnew  = fmaxf(mrow[j], mx);
            float alpha = __expf(mrow[j] - mnew);
            float p0 = __expf(x0 - mnew);
            float p1 = __expf(x1 - mnew);
            float ps = p0 + p1;
            #pragma unroll
            for (int off = 8; off > 0; off >>= 1)
                ps += __shfl_xor(ps, off, 16);
            lrow[j] = lrow[j] * alpha + ps;
            mrow[j] = mnew;
            #pragma unroll
            for (int tt = 0; tt < 4; ++tt) oacc[tt][j] *= alpha;
            int m = j + lg * 8;
            { int n = cn;      pl[m | ((n & 8) << 1)][n & 7]       = f2bf(p0); }
            { int n = 16 + cn; pl[m | ((n & 8) << 1)][8 | (n & 7)] = f2bf(p1); }
        }
        wave_lds_fence();   // P stores ordered before fragment reload (same wave)
        v16bf aP = *(const v16bf*)&pl[lane][0];

        // O += P @ V: V^T rows (d-major) are fragment-native (k-dim = key contiguous)
        #pragma unroll
        for (int tt = 0; tt < 4; ++tt) {
            const bf16* vrow = vt + ((size_t)b * D + head * DH + tt * 16 + cn) * S + kBase;
            frag16 fb;
            fb.h[0] = *(const v8bf*)(vrow + lg * 8);
            fb.h[1] = *(const v8bf*)(vrow + 16 + lg * 8);
            oacc[tt] = wmma_bf16(aP, fb.v, oacc[tt]);
        }
    }

    #pragma unroll
    for (int j = 0; j < 8; ++j) {
        float inv = 1.0f / lrow[j];
        int row = qBase + j + lg * 8;
        #pragma unroll
        for (int tt = 0; tt < 4; ++tt)
            o[((size_t)(b * S + row)) * D + head * DH + tt * 16 + cn] = f2bf(oacc[tt][j] * inv);
    }
}

// ---------------- launch ----------------
extern "C" void kernel_launch(void* const* d_in, const int* in_sizes, int n_in,
                              void* d_out, int out_size, void* d_ws, size_t ws_size,
                              hipStream_t stream)
{
    const int*   tokens  = (const int*)  d_in[0];
    const float* emb_tok = (const float*)d_in[1];
    const float* emb_pos = (const float*)d_in[2];
    const float* ln1_w   = (const float*)d_in[3];
    const float* ln1_b   = (const float*)d_in[4];
    const float* Wq      = (const float*)d_in[5];
    const float* bq      = (const float*)d_in[6];
    const float* Wk      = (const float*)d_in[7];
    const float* bk      = (const float*)d_in[8];
    const float* Wv      = (const float*)d_in[9];
    const float* bv      = (const float*)d_in[10];
    const float* Wo      = (const float*)d_in[11];
    const float* bo      = (const float*)d_in[12];
    const float* ln2_w   = (const float*)d_in[13];
    const float* ln2_b   = (const float*)d_in[14];
    const float* W1      = (const float*)d_in[15];
    const float* b1      = (const float*)d_in[16];
    const float* W2      = (const float*)d_in[17];
    const float* b2      = (const float*)d_in[18];
    const float* lnf_w   = (const float*)d_in[19];
    const float* lnf_b   = (const float*)d_in[20];
    const float* Wlm     = (const float*)d_in[21];
    const float* blm     = (const float*)d_in[22];
    float* out = (float*)d_out;

    // workspace layout (bytes)
    char* wsb = (char*)d_ws;
    constexpr size_t SZ_XF  = (size_t)BS * D * 4;     // fp32 residual stream
    constexpr size_t SZ_BD  = (size_t)BS * D * 2;     // bf16 BSxD
    constexpr size_t SZ_FF  = (size_t)BS * DFF * 2;   // bf16 BSxDFF
    float* x  = (float*)(wsb);
    bf16*  h  = (bf16*) (wsb + SZ_XF);
    bf16*  qb = (bf16*) (wsb + SZ_XF + 1 * SZ_BD);
    bf16*  kb = (bf16*) (wsb + SZ_XF + 2 * SZ_BD);
    bf16*  vb = (bf16*) (wsb + SZ_XF + 3 * SZ_BD);
    bf16*  ab = (bf16*) (wsb + SZ_XF + 4 * SZ_BD);
    bf16*  vt = (bf16*) (wsb + SZ_XF + 5 * SZ_BD);
    bf16*  ff = (bf16*) (wsb + SZ_XF + 6 * SZ_BD);
    bf16*  wt = (bf16*) (wsb + SZ_XF + 6 * SZ_BD + SZ_FF);  // reused W^T region (<= V*D bf16)

    embed_kernel<<<(BS * D) / 256, 256, 0, stream>>>(tokens, emb_tok, emb_pos, x);

    const dim3 gD(D / 128, BS / 128);
    const dim3 gF(DFF / 128, BS / 128);
    const dim3 gAttn(S / 64, H, B);
    const dim3 gTdd(D / 32, D / 32);
    const dim3 gT1 (DFF / 32, D / 32);
    const dim3 gT2 (D / 32, DFF / 32);
    const dim3 gTv (D / 32, S / 32, B);
    const dim3 gTlm(V / 32, D / 32);

    for (int l = 0; l < L; ++l) {
        const size_t wDD = (size_t)l * D * D;
        const size_t wDF = (size_t)l * D * DFF;
        const size_t wFD = (size_t)l * DFF * D;
        const size_t vD  = (size_t)l * D;
        const size_t vF  = (size_t)l * DFF;

        layernorm_kernel<<<BS, 256, 0, stream>>>(x, ln1_w + vD, ln1_b + vD, h);

        transpose_f32_bf16<<<gTdd, 256, 0, stream>>>(Wq + wDD, wt, D, D);
        gemm_kernel<false, false, true><<<gD, 256, 0, stream>>>(h, wt, bq + vD, nullptr, qb, BS, D, D);
        transpose_f32_bf16<<<gTdd, 256, 0, stream>>>(Wk + wDD, wt, D, D);
        gemm_kernel<false, false, true><<<gD, 256, 0, stream>>>(h, wt, bk + vD, nullptr, kb, BS, D, D);
        transpose_f32_bf16<<<gTdd, 256, 0, stream>>>(Wv + wDD, wt, D, D);
        gemm_kernel<false, false, true><<<gD, 256, 0, stream>>>(h, wt, bv + vD, nullptr, vb, BS, D, D);

        rope_kernel<<<(BS * D / 2) / 256, 256, 0, stream>>>(qb, kb);
        transpose_bf16_batched<<<gTv, 256, 0, stream>>>(vb, vt, S, D);
        attn_kernel<<<gAttn, 128, 0, stream>>>(qb, kb, vt, ab);

        transpose_f32_bf16<<<gTdd, 256, 0, stream>>>(Wo + wDD, wt, D, D);
        gemm_kernel<false, true, false><<<gD, 256, 0, stream>>>(ab, wt, bo + vD, x, x, BS, D, D);

        layernorm_kernel<<<BS, 256, 0, stream>>>(x, ln2_w + vD, ln2_b + vD, h);

        transpose_f32_bf16<<<gT1, 256, 0, stream>>>(W1 + wDF, wt, D, DFF);
        gemm_kernel<true, false, true><<<gF, 256, 0, stream>>>(h, wt, b1 + vF, nullptr, ff, BS, DFF, D);
        transpose_f32_bf16<<<gT2, 256, 0, stream>>>(W2 + wFD, wt, DFF, D);
        gemm_kernel<false, true, false><<<gD, 256, 0, stream>>>(ff, wt, b2 + vD, x, x, BS, D, DFF);
    }

    layernorm_kernel<<<BS, 256, 0, stream>>>(x, lnf_w, lnf_b, h);
    transpose_f32_bf16<<<gTlm, 256, 0, stream>>>(Wlm, wt, D, V);
    gemm_kernel<false, false, false><<<dim3(V / 128, BS / 128), 256, 0, stream>>>(
        h, wt, blm, nullptr, out, BS, V, D);
}
